// GroupedQueryAttention_41523743818289
// MI455X (gfx1250) — matrix operand order repaired and checked by reading, not compiled, not run on hardware
//
#include <hip/hip_runtime.h>
#include <hip/hip_bf16.h>
#include <math.h>

typedef _Float16 v16h __attribute__((ext_vector_type(16)));
typedef _Float16 v8h  __attribute__((ext_vector_type(8)));
typedef float    v8f  __attribute__((ext_vector_type(8)));

constexpr int kB   = 4;
constexpr int kS   = 1024;
constexpr int kHQ  = 32;
constexpr int kHKV = 8;
constexpr int kG   = kHQ / kHKV;   // 4
constexpr int kDK  = 128;
constexpr int kDV  = 128;
constexpr int kHID = 4096;
constexpr int kCHK = kHKV * kDK;   // 1024  K row stride (halves)
constexpr int kCHO = kHQ * kDV;    // 4096  attn row stride == GEMM contraction dim

__device__ __forceinline__ v16h pack16(v8h lo, v8h hi) {
  v16h a;
  #pragma unroll
  for (int j = 0; j < 8; ++j) { a[j] = lo[j]; a[j + 8] = hi[j]; }
  return a;
}

// ---------------------------------------------------------------- conversions
__global__ void cvt8_kernel(const float* __restrict__ src, _Float16* __restrict__ dst, int n) {
  int i = (blockIdx.x * blockDim.x + threadIdx.x) * 8;
  if (i + 7 < n) {
    const float4 a = *(const float4*)(src + i);
    const float4 b = *(const float4*)(src + i + 4);
    v8h o;
    o[0] = (_Float16)a.x; o[1] = (_Float16)a.y; o[2] = (_Float16)a.z; o[3] = (_Float16)a.w;
    o[4] = (_Float16)b.x; o[5] = (_Float16)b.y; o[6] = (_Float16)b.z; o[7] = (_Float16)b.w;
    *(v8h*)(dst + i) = o;
  }
}

// Vt[((b*HKV+h)*DV+d)*S + s] = V[((b*S+s)*HKV+h)*DV + d]   (f32 -> f16)
__global__ void vtrans_kernel(const float* __restrict__ V, _Float16* __restrict__ Vt) {
  int i  = blockIdx.x * blockDim.x + threadIdx.x;
  int s  = i & (kS - 1);
  int r  = i >> 10;
  int d  = r & (kDV - 1);
  int hb = r >> 7;
  int h  = hb & (kHKV - 1);
  int b  = hb >> 3;
  Vt[i] = (_Float16)V[((size_t)(b * kS + s) * kHKV + h) * kDV + d];
}

// ---------------------------------------------------------------- flash attention
// one wave handles one (b, head, 16-query tile); 4 waves per block
__global__ void __launch_bounds__(128)
flash_kernel(const _Float16* __restrict__ Qh,   // [B,S,HQ*DK]  (original layout, f16)
             const _Float16* __restrict__ Kh,   // [B,S,HKV*DK] (original layout, f16)
             const _Float16* __restrict__ Vt,   // [B,HKV,DV,S] (transposed, f16)
             _Float16* __restrict__ attnh)      // [B*S, HQ*DV] f16
{
  __shared__ _Float16 lds_p[4][16 * 32];        // per-wave 16x32 P tile

  const int wave = threadIdx.x >> 5;
  const int lane = threadIdx.x & 31;
  const int hh   = lane >> 4;    // half-wave group
  const int n    = lane & 15;

  const int QT  = kS / 16;                       // 64 query tiles per head
  int wid = blockIdx.x * 4 + wave;               // 0 .. 8191
  int qt  = wid % QT;
  int bh  = wid / QT;                            // b*HQ + h
  int h   = bh % kHQ;
  int b   = bh / kHQ;
  int kv  = h / kG;
  int q0  = qt * 16;

  // ---- Q A-fragments (16x32 each, d in 4 chunks of 32), held in registers
  v16h aq[4];
  {
    const _Float16* qrow = Qh + ((size_t)(b * kS + q0 + n) * kHQ + h) * kDK;
    #pragma unroll
    for (int dc = 0; dc < 4; ++dc)
      aq[dc] = pack16(*(const v8h*)(qrow + dc * 32 + hh * 16),
                      *(const v8h*)(qrow + dc * 32 + hh * 16 + 8));
  }

  v8f zero = {};
  v8f acc[8];                                    // 16 x 128 f32 output accumulator
  #pragma unroll
  for (int c = 0; c < 8; ++c) acc[c] = zero;
  float mrow[8], lrow[8];
  #pragma unroll
  for (int r = 0; r < 8; ++r) { mrow[r] = -INFINITY; lrow[r] = 0.0f; }

  const float scale = 0.08838834764831845f;      // 1/sqrt(128)
  const int nkb = (q0 + 16 + 31) >> 5;           // causal: keys 0 .. q0+15

  for (int kb = 0; kb < nkb; ++kb) {
    const int k0 = kb * 32;
    const _Float16* krow0 = Kh + (size_t)(b * kS + k0 + n) * kCHK + kv * kDK;
    const _Float16* krow1 = krow0 + (size_t)16 * kCHK;
    const _Float16* vbase = Vt + (size_t)(b * kHKV + kv) * kDV * kS + k0 + hh * 16;

    // ---- batch ALL block loads up front: 8 K fragments + 8 V fragments.
    // V loads overlap the score WMMAs + softmax + LDS round-trip below.
    v16h bk[8];
    #pragma unroll
    for (int dc = 0; dc < 4; ++dc) {
      bk[2 * dc]     = *(const v16h*)(krow0 + dc * 32 + hh * 16);
      bk[2 * dc + 1] = *(const v16h*)(krow1 + dc * 32 + hh * 16);
    }
    v16h bv[8];
    #pragma unroll
    for (int vt = 0; vt < 8; ++vt)
      bv[vt] = *(const v16h*)(vbase + (size_t)(vt * 16 + n) * kS);

    if (kb + 1 < nkb) {
      __builtin_prefetch(krow0 + (size_t)32 * kCHK, 0, 0);
      __builtin_prefetch(vbase + 32, 0, 0);
    }

    // ---- scores: two 16x16 tiles (keys k0..k0+15, k0+16..k0+31)
    v8f s0 = zero, s1 = zero;
    #pragma unroll
    for (int dc = 0; dc < 4; ++dc) {
      s0 = __builtin_amdgcn_wmma_f32_16x16x32_f16(false, aq[dc], false, bk[2 * dc],     (short)0, s0, false, false);
      s1 = __builtin_amdgcn_wmma_f32_16x16x32_f16(false, aq[dc], false, bk[2 * dc + 1], (short)0, s1, false, false);
    }

    // ---- online softmax (row = r + 8*hh lives across the 16 lanes of this half-wave)
    float bmax[8];
    #pragma unroll
    for (int r = 0; r < 8; ++r) {
      int sq = q0 + r + 8 * hh;
      float x0 = s0[r] * scale; if (k0 + n      > sq) x0 = -INFINITY;
      float x1 = s1[r] * scale; if (k0 + 16 + n > sq) x1 = -INFINITY;
      s0[r] = x0; s1[r] = x1;
      float t = fmaxf(x0, x1);
      t = fmaxf(t, __shfl_xor(t, 1, 32));
      t = fmaxf(t, __shfl_xor(t, 2, 32));
      t = fmaxf(t, __shfl_xor(t, 4, 32));
      t = fmaxf(t, __shfl_xor(t, 8, 32));
      bmax[r] = t;
    }
    #pragma unroll
    for (int r = 0; r < 8; ++r) {
      float mn    = fmaxf(mrow[r], bmax[r]);
      float alpha = __expf(mrow[r] - mn);
      mrow[r] = mn;
      float p0 = __expf(s0[r] - mn);
      float p1 = __expf(s1[r] - mn);
      float ps = p0 + p1;
      ps += __shfl_xor(ps, 1, 32);
      ps += __shfl_xor(ps, 2, 32);
      ps += __shfl_xor(ps, 4, 32);
      ps += __shfl_xor(ps, 8, 32);
      lrow[r] = lrow[r] * alpha + ps;
      #pragma unroll
      for (int c = 0; c < 8; ++c) acc[c][r] = acc[c][r] * alpha;
      // stage P (C layout -> LDS row-major 16x32)
      lds_p[wave][(r + 8 * hh) * 32 + n]      = (_Float16)p0;
      lds_p[wave][(r + 8 * hh) * 32 + 16 + n] = (_Float16)p1;
    }

    // LDS is in-order within a wave; fence the compiler + wait out stores
    asm volatile("s_wait_dscnt 0" ::: "memory");

    // ---- reload P as a 16x32 A-fragment
    v16h ap;
    {
      const _Float16* prow = &lds_p[wave][n * 32];
      ap = pack16(*(const v8h*)(prow + hh * 16), *(const v8h*)(prow + hh * 16 + 8));
    }

    // ---- PV: acc += P(16x32) x V(32x16) over 8 column tiles of DV
    #pragma unroll
    for (int vt = 0; vt < 8; ++vt)
      acc[vt] = __builtin_amdgcn_wmma_f32_16x16x32_f16(false, ap, false, bv[vt], (short)0, acc[vt], false, false);
  }

  // ---- normalize and write attn output (f16, row-major [B*S, HQ*DV])
  _Float16* orow = attnh + (size_t)(b * kS + q0) * kCHO + h * kDV;
  #pragma unroll
  for (int r = 0; r < 8; ++r) {
    float inv = 1.0f / lrow[r];
    _Float16* prow = orow + (size_t)(r + 8 * hh) * kCHO;
    #pragma unroll
    for (int vt = 0; vt < 8; ++vt)
      prow[vt * 16 + n] = (_Float16)(acc[vt][r] * inv);
  }
}

// ---------------------------------------------------------------- output projection
// out[t, hid] = bias[hid] + sum_c attn[t,c] * W_o[hid,c]
// wave tile: 32 rows x 64 cols; 4 waves/block -> 32 x 256 block tile
// K loop is software-pipelined (double-buffered fragments).
__global__ void __launch_bounds__(128)
gemm_kernel(const _Float16* __restrict__ A,    // [4096, 4096]  attn (f16)
            const _Float16* __restrict__ Wh,   // [4096, 4096]  W_o (f16, row-major)
            const float* __restrict__ bias,    // [4096]
            float* __restrict__ out)           // [4096, 4096]  f32
{
  const int wave = threadIdx.x >> 5;
  const int lane = threadIdx.x & 31;
  const int hh   = lane >> 4;
  const int n    = lane & 15;

  const int t0 = blockIdx.x * 32;                    // row tile base
  const int c0 = blockIdx.y * 256 + wave * 64;       // col tile base
  constexpr int KC = kCHO / 32;                      // 128 K-steps

  v8f zero = {};
  v8f acc[2][4];
  #pragma unroll
  for (int rt = 0; rt < 2; ++rt)
    #pragma unroll
    for (int ct = 0; ct < 4; ++ct) acc[rt][ct] = zero;

  auto loadFrag = [&](int kc, v16h (&aA)[2], v16h (&bB)[4]) {
    #pragma unroll
    for (int rt = 0; rt < 2; ++rt) {
      const _Float16* ar = A + (size_t)(t0 + rt * 16 + n) * kCHO + kc * 32;
      aA[rt] = pack16(*(const v8h*)(ar + hh * 16), *(const v8h*)(ar + hh * 16 + 8));
    }
    #pragma unroll
    for (int ct = 0; ct < 4; ++ct)
      bB[ct] = *(const v16h*)(Wh + (size_t)(c0 + ct * 16 + n) * kCHO + kc * 32 + hh * 16);
  };
  auto compute = [&](v16h (&aA)[2], v16h (&bB)[4]) {
    #pragma unroll
    for (int ct = 0; ct < 4; ++ct)
      #pragma unroll
      for (int rt = 0; rt < 2; ++rt)
        acc[rt][ct] = __builtin_amdgcn_wmma_f32_16x16x32_f16(false, aA[rt], false, bB[ct], (short)0,
                                                             acc[rt][ct], false, false);
  };

  v16h a0[2], b0v[4], a1[2], b1v[4];
  loadFrag(0, a0, b0v);
  for (int kc = 0; kc < KC; kc += 2) {
    loadFrag(kc + 1, a1, b1v);     // prefetch odd step while computing even
    compute(a0, b0v);
    if (kc + 2 < KC) loadFrag(kc + 2, a0, b0v);  // prefetch next even step
    compute(a1, b1v);
  }

  #pragma unroll
  for (int ct = 0; ct < 4; ++ct) {
    float bz = bias[c0 + ct * 16 + n];
    #pragma unroll
    for (int rt = 0; rt < 2; ++rt)
      #pragma unroll
      for (int r = 0; r < 8; ++r)
        out[(size_t)(t0 + rt * 16 + r + 8 * hh) * kHID + c0 + ct * 16 + n] = acc[rt][ct][r] + bz;
  }
}

// ---------------------------------------------------------------- launch
extern "C" void kernel_launch(void* const* d_in, const int* in_sizes, int n_in,
                              void* d_out, int out_size, void* d_ws, size_t ws_size,
                              hipStream_t stream) {
  const float* Q   = (const float*)d_in[0];
  const float* K   = (const float*)d_in[1];
  const float* V   = (const float*)d_in[2];
  // d_in[3] = mask (tril causal) -> applied analytically in flash_kernel
  const float* W   = (const float*)d_in[4];
  const float* b_o = (const float*)d_in[5];
  float* out = (float*)d_out;

  const size_t nQ = (size_t)kB * kS * kHQ * kDK;   // 16M
  const size_t nK = (size_t)kB * kS * kHKV * kDK;  //  4M
  const size_t nV = (size_t)kB * kS * kHKV * kDV;  //  4M
  const size_t nW = (size_t)kHID * kCHO;           // 16M

  _Float16* Qh = (_Float16*)d_ws;
  _Float16* Kh = Qh + nQ;
  _Float16* Vt = Kh + nK;
  _Float16* Wh = Vt + nV;
  _Float16* Ah = Wh + nW;

  cvt8_kernel<<<(int)(nQ / 8 / 256), 256, 0, stream>>>(Q, Qh, (int)nQ);
  cvt8_kernel<<<(int)(nK / 8 / 256), 256, 0, stream>>>(K, Kh, (int)nK);
  cvt8_kernel<<<(int)(nW / 8 / 256), 256, 0, stream>>>(W, Wh, (int)nW);
  vtrans_kernel<<<(int)(nV / 256), 256, 0, stream>>>(V, Vt);

  // B*HQ*(S/16) = 8192 waves, 4 waves/block
  flash_kernel<<<2048, 128, 0, stream>>>(Qh, Kh, Vt, Ah);

  // 4096x4096x4096 GEMM: grid = (rows/32, cols/256)
  dim3 ggrid((kB * kS) / 32, kHID / 256);
  gemm_kernel<<<ggrid, 128, 0, stream>>>(Ah, Wh, b_o, out);
}
